// LowRankNet_75041668596538
// MI455X (gfx1250) — matrix operand order repaired and checked by reading, not compiled, not run on hardware
//
#include <hip/hip_runtime.h>
#include <hip/hip_bf16.h>

typedef __attribute__((ext_vector_type(16))) _Float16 v16h;
typedef __attribute__((ext_vector_type(8)))  _Float16 v8h;
typedef __attribute__((ext_vector_type(8)))  float    v8f;
typedef __attribute__((ext_vector_type(4)))  float    v4f;

constexpr int kB  = 4;
constexpr int kC  = 64;
constexpr int kT  = 80;
constexpr int kH  = 36;
constexpr int kW  = 64;
constexpr int kN  = 128;
constexpr int kNT = 10;
constexpr int kS  = kH * kW;          // 2304 = 72 * 32
constexpr int kSH = kS / 2;           // 1152 (per-wave K half)
constexpr int kTOUT = kT - kNT + 1;   // 71

// ---------------------------------------------------------------------------
// Kernel 1: Gaussian soft-pool weights, normalized per target, stored
// TRANSPOSED in f16 (wsT[n][s]) so WMMA B-fragments are contiguous.
// ---------------------------------------------------------------------------
__global__ void lrn_ws_kernel(const float* __restrict__ wx,
                              const float* __restrict__ wy,
                              const float* __restrict__ wsigmax,
                              const float* __restrict__ wsigmay,
                              _Float16* __restrict__ wsT) {
  __shared__ float red[256];
  const int n   = blockIdx.x;
  const int tid = threadIdx.x;

  const float rx = fmaxf(wsigmax[n], 0.0f);
  const float ry = fmaxf(wsigmay[n], 0.0f);
  const float inv2sx = 1.0f / (2.0f * (0.1f + rx * rx));
  const float inv2sy = 1.0f / (2.0f * (0.1f + ry * ry));
  const float cx = wx[n];
  const float cy = wy[n];

  float e[9];
  float lsum = 0.0f;
#pragma unroll
  for (int i = 0; i < 9; ++i) {
    const int s  = tid + i * 256;
    const int hi = s / kW;
    const int wi = s % kW;
    const float xv = ((float)wi - (float)kW * 0.5f + 0.5f) / (float)kW;
    const float yv = ((float)hi - (float)kH * 0.5f + 0.5f) / (float)kH;
    const float dx = xv - cx;
    const float dy = yv - cy;
    e[i] = expf(-(dx * dx * inv2sx + dy * dy * inv2sy));
    lsum += e[i];
  }

  red[tid] = lsum;
  __syncthreads();
  for (int off = 128; off > 0; off >>= 1) {
    if (tid < off) red[tid] += red[tid + off];
    __syncthreads();
  }
  const float inv = 1.0f / (0.1f + red[0]);

#pragma unroll
  for (int i = 0; i < 9; ++i) {
    const int s = tid + i * 256;
    wsT[(size_t)n * kS + s] = (_Float16)(e[i] * inv);
  }
}

// ---------------------------------------------------------------------------
// Kernel 2: per-frame GEMM, f16 WMMA / fp32 accumulate, B staged in LDS.
// Block = one frame f; 8 waves: wave w -> row-tile (w&3), K-half (w>>2),
// all 8 column tiles. A is read exactly once (nontemporal); B chunks are
// cooperatively staged into LDS (conflict-free padded layout) and reused by
// all waves. K per wave = 1152 in 18 chunks of 64 (2 WMMA k-steps each).
// sched_group_barriers pin the pipeline per k-step: 16 DS reads -> 8 WMMAs.
// ---------------------------------------------------------------------------
constexpr int KC  = 64;        // staged K-chunk per half (halves)
constexpr int STR = KC + 8;    // padded LDS row stride (halves); 36 DWORDs
constexpr int NCHUNK = kSH / KC;  // 18

__global__ void __launch_bounds__(256)
lrn_gemm_kernel(const float* __restrict__ x,
                const _Float16* __restrict__ wsT,
                const float* __restrict__ wc,
                float* __restrict__ R) {
  __shared__ _Float16 ldsb[2 * kN * STR];   // 2 halves * 128 n * 72 = 36.9 KB
  __shared__ float Rpart[8][kN];

  const int f = blockIdx.x;
  const int b = f / kT;
  const int t = f % kT;

  const int tid    = threadIdx.x;
  const int lane   = tid & 31;
  const int wave   = tid >> 5;
  const int rt     = wave & 3;    // row tile: channels [16*rt, 16*rt+15]
  const int kh     = wave >> 2;   // K half: [kh*1152, kh*1152+1151]
  const int lrow   = lane & 15;
  const bool hilane = lane >= 16;
  const int kbaseA = hilane ? 8 : 0;   // A layout: lo lanes K[0..7],K[16..23]
  const int kbaseB = hilane ? 8 : 0;   // B layout: lo lanes K[0..7],K[16..23]

  // This lane's A row (channel c), offset to its K half; contiguous in s.
  const int c = rt * 16 + lrow;
  const float* arow =
      x + (size_t)(((b * kC + c) * kT) + t) * kS + (size_t)kh * kSH;

  v8f acc[8] = {v8f{}, v8f{}, v8f{}, v8f{}, v8f{}, v8f{}, v8f{}, v8f{}};

  for (int ch = 0; ch < NCHUNK; ++ch) {
    __syncthreads();  // previous chunk fully consumed

    // ---- cooperative stage of both K-half sub-chunks: 2*128*64 halves ----
    // 8-half (16B) segments; 2048 segments / 256 threads = 8 per thread.
#pragma unroll
    for (int i = 0; i < 8; ++i) {
      const int seg = tid + i * 256;        // 0..2047
      const int sub = seg >> 10;            // which K half
      const int n   = (seg >> 3) & 127;     // target row
      const int kp  = (seg & 7) * 8;        // halves within chunk
      const _Float16* src =
          wsT + (size_t)n * kS + (size_t)sub * kSH + ch * KC + kp;
      *(v4f*)&ldsb[(sub * kN + n) * STR + kp] = *(const v4f*)src;
    }
    __syncthreads();

    // ---- compute KC/32 = 2 WMMA k-steps on the staged chunk ----
#pragma unroll
    for (int ks = 0; ks < KC / 32; ++ks) {
      const int kg = ch * KC + ks * 32;     // within this wave's K half

      // A fragment: fp32 -> f16 into CDNA5 16x32 A layout (streamed, NT).
      const float* ap = arow + kg + kbaseA;
      const v4f p0 = __builtin_nontemporal_load((const v4f*)(ap));
      const v4f p1 = __builtin_nontemporal_load((const v4f*)(ap + 4));
      const v4f p2 = __builtin_nontemporal_load((const v4f*)(ap + 16));
      const v4f p3 = __builtin_nontemporal_load((const v4f*)(ap + 20));
      v16h a;
      a[0]  = (_Float16)p0[0]; a[1]  = (_Float16)p0[1];
      a[2]  = (_Float16)p0[2]; a[3]  = (_Float16)p0[3];
      a[4]  = (_Float16)p1[0]; a[5]  = (_Float16)p1[1];
      a[6]  = (_Float16)p1[2]; a[7]  = (_Float16)p1[3];
      a[8]  = (_Float16)p2[0]; a[9]  = (_Float16)p2[1];
      a[10] = (_Float16)p2[2]; a[11] = (_Float16)p2[3];
      a[12] = (_Float16)p3[0]; a[13] = (_Float16)p3[1];
      a[14] = (_Float16)p3[2]; a[15] = (_Float16)p3[3];

      // Batch-load ALL 8 B fragments into distinct registers.
      const _Float16* lb = &ldsb[(kh * kN) * STR + ks * 32 + kbaseB];
      v16h bf[8];
#pragma unroll
      for (int j = 0; j < 8; ++j) {
        const _Float16* bp = lb + (j * 16 + lrow) * STR;
        const v8h blo = *(const v8h*)(bp);
        const v8h bhi = *(const v8h*)(bp + 16);
        bf[j] = __builtin_shufflevector(
            blo, bhi, 0, 1, 2, 3, 4, 5, 6, 7, 8, 9, 10, 11, 12, 13, 14, 15);
      }

      // 8 independent WMMAs back-to-back (same A, distinct B/acc).
#pragma unroll
      for (int j = 0; j < 8; ++j) {
        acc[j] = __builtin_amdgcn_wmma_f32_16x16x32_f16(
            false, a, false, bf[j], (short)0, acc[j], false, false);
      }

      // Pipeline directive for this k-step: issue all 16 DS reads first,
      // then the 8 WMMAs (waitcnt pass can stagger dscnt waits).
      __builtin_amdgcn_sched_group_barrier(0x100, 16, 0);  // DS reads
      __builtin_amdgcn_sched_group_barrier(0x008, 8, 0);   // WMMA
    }
  }

  // ---- wc-weighted channel reduction of D tiles ----
  // D layout: VGPR v, lane L -> M = v + (L>=16 ? 8 : 0), N = L%16.
  const int cbase = rt * 16 + (hilane ? 8 : 0);
#pragma unroll
  for (int j = 0; j < 8; ++j) {
    const int n = j * 16 + lrow;
    float p = 0.0f;
#pragma unroll
    for (int v = 0; v < 8; ++v) {
      p += acc[j][v] * wc[(cbase + v) * kN + n];
    }
    p += __shfl_xor(p, 16, 32);   // fold the two M-halves (same n)
    if (!hilane) Rpart[wave][n] = p;
  }
  __syncthreads();

  if (tid < kN) {
    float s = 0.0f;
#pragma unroll
    for (int w = 0; w < 8; ++w) s += Rpart[w][tid];
    R[(size_t)f * kN + tid] = s;
  }
}

// ---------------------------------------------------------------------------
// Kernel 3: grouped temporal conv (VALID cross-correlation) + bias.
// out[b, n, t'] = sum_k R[b*T + t' + k, n] * wt[k, n] + wb[n]
// ---------------------------------------------------------------------------
__global__ void lrn_conv_kernel(const float* __restrict__ R,
                                const float* __restrict__ wt,
                                const float* __restrict__ wb,
                                float* __restrict__ out) {
  const int idx = blockIdx.x * blockDim.x + threadIdx.x;
  if (idx >= kB * kN * kTOUT) return;
  const int tp = idx % kTOUT;
  const int n  = (idx / kTOUT) % kN;
  const int b  = idx / (kTOUT * kN);

  float s = wb[n];
#pragma unroll
  for (int k = 0; k < kNT; ++k) {
    s += R[(size_t)(b * kT + tp + k) * kN + n] * wt[k * kN + n];
  }
  out[idx] = s;
}

// ---------------------------------------------------------------------------
// Launch: inputs in dict order:
// 0:x 1:targets(unused) 2:wc 3:wx 4:wy 5:wsigmax 6:wsigmay 7:wt 8:wb
// ---------------------------------------------------------------------------
extern "C" void kernel_launch(void* const* d_in, const int* in_sizes, int n_in,
                              void* d_out, int out_size, void* d_ws, size_t ws_size,
                              hipStream_t stream) {
  const float* x       = (const float*)d_in[0];
  const float* wc      = (const float*)d_in[2];
  const float* wx      = (const float*)d_in[3];
  const float* wy      = (const float*)d_in[4];
  const float* wsigmax = (const float*)d_in[5];
  const float* wsigmay = (const float*)d_in[6];
  const float* wt      = (const float*)d_in[7];
  const float* wb      = (const float*)d_in[8];
  float* out = (float*)d_out;

  // Scratch layout: R fp32 [320*128] then wsT f16 [128*2304] (256B aligned).
  char* wsbase = (char*)d_ws;
  float*    R   = (float*)wsbase;                 // 163840 bytes
  _Float16* wsT = (_Float16*)(wsbase + 164096);   // 589824 bytes

  lrn_ws_kernel<<<kN, 256, 0, stream>>>(wx, wy, wsigmax, wsigmay, wsT);
  lrn_gemm_kernel<<<kB * kT, 256, 0, stream>>>(x, wsT, wc, R);
  lrn_conv_kernel<<<(kB * kN * kTOUT + 255) / 256, 256, 0, stream>>>(R, wt, wb, out);
}
